// DL_Loop_Model_v20230613_25168508355234
// MI455X (gfx1250) — compile-verified, matrix operand order checked by reading, not run on hardware
//
#include <hip/hip_runtime.h>
#include <hip/hip_bf16.h>
#include <math.h>

// ---------------------------------------------------------------------------
// Types for CDNA5 WMMA
// ---------------------------------------------------------------------------
typedef __attribute__((ext_vector_type(16))) _Float16 v16h;
typedef __attribute__((ext_vector_type(8)))  _Float16 v8h;
typedef __attribute__((ext_vector_type(8)))  float    v8f;

#define B_ROWS   2048
#define N7       78
#define N3       121
#define NBR      (N7 + N3)     // 199
#define GRU_H    16
#define EPS_BN   1e-5f

#define MLP_WAVES 4            // waves per MLP block
#define MLP_TPW   4            // 16-row batch tiles per wave
#define MLP_BLOCKS (B_ROWS / (16 * MLP_WAVES * MLP_TPW))   // 8

__device__ __forceinline__ float sigmoidf_(float x) {
  return 1.0f / (1.0f + __expf(-x));
}

// ---------------------------------------------------------------------------
// Fragment loader: 16x32 f16 fragment (A layout, ISA 7.12.2). Per lane the 16
// halves live in exactly two contiguous 8-half runs of the source row:
//   lanes 0-15 : K = [0..7]  and [16..23]
//   lanes 16-31: K = [8..15] and [24..31]
// -> two ds_load_b128 per lane. Used for A (row = batch row) and B (row =
// output column of an LDS-transposed weight tile).
// ---------------------------------------------------------------------------
__device__ __forceinline__ v16h loadFrag(const _Float16* lds, int stride,
                                         int kOff, int row, int lane) {
  int base = (lane < 16) ? 0 : 8;
  const _Float16* p0 = lds + row * stride + kOff + base;
  v8h lo = *(const v8h*)p0;
  v8h hi = *(const v8h*)(p0 + 16);
  return __builtin_shufflevector(lo, hi, 0, 1, 2, 3, 4, 5, 6, 7,
                                 8, 9, 10, 11, 12, 13, 14, 15);
}

// Stage global f32 weights W [K][NTOT] -> LDS transposed f16 Wt [NTOT][KPAD],
// zero-padded along K. Cooperatively over the whole block (STEP threads).
template <int KVALID, int KPAD, int NTOT, int STEP>
__device__ __forceinline__ void stageW(const float* __restrict__ W,
                                       _Float16* __restrict__ Wt, int tid) {
  for (int i = tid; i < NTOT * KPAD; i += STEP) {
    int n = i / KPAD, k = i % KPAD;
    float v = (k < KVALID) ? W[k * NTOT + n] : 0.0f;
    Wt[n * KPAD + k] = (_Float16)v;
  }
}

// ---------------------------------------------------------------------------
// Kernel 1: one-hot(5) + raw features -> f19T [19][2048]  (transposed so the
// MLP staging reads one contiguous 64B run per lane)
// ---------------------------------------------------------------------------
__global__ __launch_bounds__(256)
void f19_kernel(const float* __restrict__ features, float* __restrict__ f19T) {
  int b = blockIdx.x * blockDim.x + threadIdx.x;
  if (b >= B_ROWS) return;
  int cls = (int)features[b * 14];
  cls = cls < 0 ? 0 : (cls > 4 ? 4 : cls);
#pragma unroll
  for (int j = 0; j < 5; ++j) f19T[j * B_ROWS + b] = (j == cls) ? 1.0f : 0.0f;
#pragma unroll
  for (int j = 0; j < 14; ++j) f19T[(5 + j) * B_ROWS + b] = features[b * 14 + j];
}

// ---------------------------------------------------------------------------
// Kernel 2: per-branch MLP via WMMA. 4 waves/block; all three weight tiles
// staged once (transposed f16, zero-padded) in LDS; BN folded to scale/shift
// registers; each wave sweeps 4 private 16-row batch tiles barrier-free.
// Output: h32 activations, f16, layout [branch][t][32] for GRU streaming.
// ---------------------------------------------------------------------------
struct BranchMlpParams {
  const int* idx;
  int branchBase;
  const float *bn0_g, *bn0_b, *bn0_m, *bn0_v;
  const float *W0, *b0, *W1, *b1;
  const float *bn1_g, *bn1_b, *bn1_m, *bn1_v;
  const float *W2, *b2;
  const float *bn2_g, *bn2_b, *bn2_m, *bn2_v;
};

template <int FVAL>
__global__ __launch_bounds__(32 * MLP_WAVES)
void branch_mlp_kernel(BranchMlpParams P, const float* __restrict__ f19T,
                       _Float16* __restrict__ h32out) {
  __shared__ __align__(16) _Float16 W0t[32 * 32];            // [n][kpad=32]
  __shared__ __align__(16) _Float16 W1t[64 * 32];            // [n][kpad=32]
  __shared__ __align__(16) _Float16 W2t[32 * 64];            // [n][kpad=64]
  __shared__ __align__(16) _Float16 actA[MLP_WAVES][16 * 64];
  __shared__ __align__(16) _Float16 actB[MLP_WAVES][16 * 64];

  const int tid  = threadIdx.x;
  const int lane = tid & 31;
  const int wid  = tid >> 5;
  const int nb   = blockIdx.y;            // local branch
  const int gb   = P.branchBase + nb;     // global branch id
  const int l16  = lane & 15;

  // ---- stage all weight tiles once per block ----
  stageW<FVAL, 32, 32, 32 * MLP_WAVES>(P.W0 + (size_t)nb * FVAL * 32, W0t, tid);
  stageW<32, 32, 64, 32 * MLP_WAVES>(P.W1 + (size_t)nb * 32 * 64, W1t, tid);
  stageW<64, 64, 32, 32 * MLP_WAVES>(P.W2 + (size_t)nb * 64 * 32, W2t, tid);

  // ---- per-lane loop-invariant constants ----
  // BN0: in the 16-step staging loop, column k == lane.
  float scale0 = 0.0f, shift0 = 0.0f;
  const float* colp = f19T;               // gather column base (lane < FVAL)
  if (lane < FVAL) {
    int col = P.idx[nb * FVAL + lane];
    float g = P.bn0_g[nb * FVAL + lane], b = P.bn0_b[nb * FVAL + lane];
    float m = P.bn0_m[nb * FVAL + lane], v = P.bn0_v[nb * FVAL + lane];
    scale0 = g * __frsqrt_rn(v + EPS_BN);
    shift0 = b - m * scale0;
    colp   = f19T + (size_t)col * B_ROWS;
  }
  // Layer biases / folded BN (channel n = l16 + 16*nt)
  float c0[2], s1[4], o1[4], s2[2], o2[2];
#pragma unroll
  for (int nt = 0; nt < 2; ++nt) c0[nt] = P.b0[nb * 32 + l16 + 16 * nt];
#pragma unroll
  for (int nt = 0; nt < 4; ++nt) {
    int n = l16 + 16 * nt;
    float sc = P.bn1_g[nb * 64 + n] * __frsqrt_rn(P.bn1_v[nb * 64 + n] + EPS_BN);
    s1[nt] = sc;
    o1[nt] = (P.b1[nb * 64 + n] - P.bn1_m[nb * 64 + n]) * sc + P.bn1_b[nb * 64 + n];
  }
#pragma unroll
  for (int nt = 0; nt < 2; ++nt) {
    int n = l16 + 16 * nt;
    float sc = P.bn2_g[nb * 32 + n] * __frsqrt_rn(P.bn2_v[nb * 32 + n] + EPS_BN);
    s2[nt] = sc;
    o2[nt] = (P.b2[nb * 32 + n] - P.bn2_m[nb * 32 + n]) * sc + P.bn2_b[nb * 32 + n];
  }
  __syncthreads();   // weights staged

  _Float16* myA = actA[wid];
  _Float16* myB = actB[wid];

  for (int it = 0; it < MLP_TPW; ++it) {
    const int tile = (blockIdx.x * MLP_WAVES + wid) * MLP_TPW + it;
    const int row0 = tile * 16;

    // ---- stage BN0(x) into myA [16][32] f16 (lane owns column k = lane) ----
#pragma unroll
    for (int m = 0; m < 16; ++m) {
      float xv = (lane < FVAL) ? colp[row0 + m] : 0.0f;
      float bn = (lane < FVAL) ? fmaf(xv, scale0, shift0) : 0.0f;
      myA[m * 32 + lane] = (_Float16)bn;
    }
    __builtin_amdgcn_wave_barrier();

    // ---- layer 0: [16,32pad] x [32pad,32] -> +b, ReLU -> myB [16][32] ----
    v16h A0 = loadFrag(myA, 32, 0, l16, lane);
#pragma unroll
    for (int nt = 0; nt < 2; ++nt) {
      v16h Bf = loadFrag(W0t, 32, 0, l16 + 16 * nt, lane);
      v8f acc = {};
      acc = __builtin_amdgcn_wmma_f32_16x16x32_f16(false, A0, false, Bf,
                                                   (short)0, acc, false, false);
#pragma unroll
      for (int r = 0; r < 8; ++r) {
        int m = r + ((lane < 16) ? 0 : 8);
        int n = l16 + 16 * nt;
        myB[m * 32 + n] = (_Float16)fmaxf(acc[r] + c0[nt], 0.0f);
      }
    }
    __builtin_amdgcn_wave_barrier();

    // ---- layer 1: [16,32] x [32,64] -> BN1+ReLU (folded) -> myA [16][64] ----
    v16h A1 = loadFrag(myB, 32, 0, l16, lane);
#pragma unroll
    for (int nt = 0; nt < 4; ++nt) {
      v16h Bf = loadFrag(W1t, 32, 0, l16 + 16 * nt, lane);
      v8f acc = {};
      acc = __builtin_amdgcn_wmma_f32_16x16x32_f16(false, A1, false, Bf,
                                                   (short)0, acc, false, false);
#pragma unroll
      for (int r = 0; r < 8; ++r) {
        int m = r + ((lane < 16) ? 0 : 8);
        int n = l16 + 16 * nt;
        myA[m * 64 + n] = (_Float16)fmaxf(fmaf(acc[r], s1[nt], o1[nt]), 0.0f);
      }
    }
    __builtin_amdgcn_wave_barrier();

    // ---- layer 2: [16,64] x [64,32], K split into 2 chained WMMAs ----
#pragma unroll
    for (int nt = 0; nt < 2; ++nt) {
      v8f acc = {};
#pragma unroll
      for (int kc = 0; kc < 2; ++kc) {
        v16h Af = loadFrag(myA, 64, 32 * kc, l16, lane);
        v16h Bf = loadFrag(W2t, 64, 32 * kc, l16 + 16 * nt, lane);
        acc = __builtin_amdgcn_wmma_f32_16x16x32_f16(false, Af, false, Bf,
                                                     (short)0, acc, false, false);
      }
#pragma unroll
      for (int r = 0; r < 8; ++r) {
        int m = r + ((lane < 16) ? 0 : 8);
        int n = l16 + 16 * nt;
        myB[m * 32 + n] = (_Float16)fmaxf(fmaf(acc[r], s2[nt], o2[nt]), 0.0f);
      }
    }
    __builtin_amdgcn_wave_barrier();

    // ---- write activations [branch][t][32] f16: lane = (row, half), 2xb128 ----
    {
      int m = l16, half = lane >> 4;
      const uint4* src = (const uint4*)(myB + m * 32 + half * 16);
      uint4 q0 = src[0], q1 = src[1];
      uint4* dst = (uint4*)(h32out + ((size_t)gb * B_ROWS + row0 + m) * 32 + half * 16);
      dst[0] = q0; dst[1] = q1;
    }
    __builtin_amdgcn_wave_barrier();
  }
}

// ---------------------------------------------------------------------------
// Kernel 3: GRU recurrence (one wave per branch x direction). Weights and the
// replicated 16-wide hidden state live in VGPRs; cross-lane traffic is only
// shuffles. Per step, also reduces hf/hb . Wout into a scalar partial.
// ---------------------------------------------------------------------------
struct GruParams {
  int branchBase;
  const float *Wih_f, *Whh_f, *bih_f, *bhh_f;
  const float *Wih_b, *Whh_b, *bih_b, *bhh_b;
  const float *Wout;             // [n][32]; fwd uses [0:16], bwd [16:32]
};

__global__ __launch_bounds__(32)
void gru_kernel(GruParams P, const _Float16* __restrict__ h32,
                float* __restrict__ partial) {
  const int lane = threadIdx.x;
  const int dir  = blockIdx.x;            // 0 = forward, 1 = backward
  const int nb   = blockIdx.y;
  const int gb   = P.branchBase + nb;
  const int l16  = lane & 15;

  const float* Wih = dir ? P.Wih_b : P.Wih_f;
  const float* Whh = dir ? P.Whh_b : P.Whh_f;
  const float* bih = dir ? P.bih_b : P.bih_f;
  const float* bhh = dir ? P.bhh_b : P.bhh_f;

  // lane g = lane holds gate row g (r/z block, rows 0..31);
  // lanes 0..15 additionally hold row 32+l (the c block).
  float wih0[32], wih1[32], whh0[16], whh1[16];
#pragma unroll
  for (int j = 0; j < 32; ++j) {
    wih0[j] = Wih[((size_t)nb * 48 + lane) * 32 + j];
    wih1[j] = Wih[((size_t)nb * 48 + 32 + l16) * 32 + j];
  }
#pragma unroll
  for (int j = 0; j < 16; ++j) {
    whh0[j] = Whh[((size_t)nb * 48 + lane) * 16 + j];
    whh1[j] = Whh[((size_t)nb * 48 + 32 + l16) * 16 + j];
  }
  const float bi0 = bih[nb * 48 + lane];
  const float bh0 = bhh[nb * 48 + lane];
  const float bi1 = bih[nb * 48 + 32 + l16];
  const float bh1 = bhh[nb * 48 + 32 + l16];
  const float wout = (lane < 16) ? P.Wout[(size_t)nb * 32 + dir * 16 + lane] : 0.0f;

  float h[16];
#pragma unroll
  for (int j = 0; j < 16; ++j) h[j] = 0.0f;

  for (int s = 0; s < B_ROWS; ++s) {
    const int tt = dir ? (B_ROWS - 1 - s) : s;
    const _Float16* xp = h32 + ((size_t)gb * B_ROWS + tt) * 32;
    // prefetch next timestep row (global_prefetch_b8, speculative)
    const int tn = dir ? (tt - 1) : (tt + 1);
    __builtin_prefetch(h32 + ((size_t)gb * B_ROWS + (tn & (B_ROWS - 1))) * 32, 0, 1);

    // 64B activation row as four b128 loads
    const v8h* xv = (const v8h*)xp;
    v8h xq0 = xv[0], xq1 = xv[1], xq2 = xv[2], xq3 = xv[3];
    float x[32];
#pragma unroll
    for (int j = 0; j < 8; ++j) {
      x[j]      = (float)xq0[j];
      x[j + 8]  = (float)xq1[j];
      x[j + 16] = (float)xq2[j];
      x[j + 24] = (float)xq3[j];
    }

    float gi0 = bi0, gi1 = bi1;
#pragma unroll
    for (int j = 0; j < 32; ++j) {
      gi0 = fmaf(wih0[j], x[j], gi0);
      gi1 = fmaf(wih1[j], x[j], gi1);
    }
    float gh0 = bh0, gh1 = bh1;
#pragma unroll
    for (int j = 0; j < 16; ++j) {
      gh0 = fmaf(whh0[j], h[j], gh0);
      gh1 = fmaf(whh1[j], h[j], gh1);
    }
    float s0 = sigmoidf_(gi0 + gh0);          // r in lanes 0-15, z in lanes 16-31
    float z  = __shfl_xor(s0, 16, 32);        // lanes 0-15 receive z
    float c  = tanhf(gi1 + s0 * gh1);         // uses r (valid in lanes 0-15)
    float hn = (1.0f - z) * c + z * h[l16];   // new hidden, lanes 0-15

    // per-timestep output dot: reduce hn*wout over lanes 0..15
    float pd = (lane < 16) ? hn * wout : 0.0f;
    pd += __shfl_xor(pd, 8, 32);
    pd += __shfl_xor(pd, 4, 32);
    pd += __shfl_xor(pd, 2, 32);
    pd += __shfl_xor(pd, 1, 32);
    if (lane == 0)
      partial[((size_t)gb * B_ROWS + tt) * 2 + dir] = pd;

    // broadcast refreshed hidden state to all lanes
#pragma unroll
    for (int j = 0; j < 16; ++j) h[j] = __shfl(hn, j, 32);
  }
}

// ---------------------------------------------------------------------------
// Kernel 4: dl[b][n] = sigmoid(partial_f + partial_b + bout[n]) -> d_out[2048+...]
// ---------------------------------------------------------------------------
__global__ __launch_bounds__(256)
void combine_kernel(const float* __restrict__ partial,
                    const float* __restrict__ bout7,
                    const float* __restrict__ bout3,
                    float* __restrict__ d_out) {
  int i = blockIdx.x * blockDim.x + threadIdx.x;
  if (i >= B_ROWS * NBR) return;
  int b = i / NBR, n = i % NBR;
  size_t pbase = ((size_t)n * B_ROWS + b) * 2;
  float p = partial[pbase] + partial[pbase + 1];
  float bo = (n < N7) ? bout7[n] : bout3[n - N7];
  d_out[B_ROWS + (size_t)b * NBR + n] = sigmoidf_(p + bo);
}

// ---------------------------------------------------------------------------
// Kernel 5: head 300->16 (ReLU) -> 1 (sigmoid). Only first 199 inputs nonzero.
// ---------------------------------------------------------------------------
__global__ __launch_bounds__(256)
void out_kernel(float* __restrict__ d_out, const float* __restrict__ W1,
                const float* __restrict__ b1, const float* __restrict__ W2,
                const float* __restrict__ b2) {
  __shared__ float sW1[NBR * 16];
  __shared__ float sb1[16], sW2[16];
  int tid = threadIdx.x;
  for (int i = tid; i < NBR * 16; i += 256) sW1[i] = W1[i];
  if (tid < 16) { sb1[tid] = b1[tid]; sW2[tid] = W2[tid]; }
  __syncthreads();
  int b = blockIdx.x * 256 + tid;
  if (b >= B_ROWS) return;
  const float* dl = d_out + B_ROWS + (size_t)b * NBR;
  float hid[16];
#pragma unroll
  for (int hh = 0; hh < 16; ++hh) hid[hh] = sb1[hh];
  for (int n = 0; n < NBR; ++n) {
    float v = dl[n];
#pragma unroll
    for (int hh = 0; hh < 16; ++hh) hid[hh] = fmaf(v, sW1[n * 16 + hh], hid[hh]);
  }
  float acc = b2[0];
#pragma unroll
  for (int hh = 0; hh < 16; ++hh) acc += fmaxf(hid[hh], 0.0f) * sW2[hh];
  d_out[b] = sigmoidf_(acc);
}

// ---------------------------------------------------------------------------
// Host-side launch
// ---------------------------------------------------------------------------
// d_in layout (top-level dict order; nested dicts in insertion order):
enum { IN_FEAT = 0, IN_IDX7 = 1, IN_IDX3 = 2, P7_BASE = 3, P3_BASE = 31, OUT_BASE = 59 };
enum {
  K_BN0G = 0, K_BN0B, K_BN0M, K_BN0V, K_W0, K_B0, K_W1, K_B1,
  K_BN1G, K_BN1B, K_BN1M, K_BN1V, K_W2, K_B2, K_BN2G, K_BN2B, K_BN2M, K_BN2V,
  K_WIHF, K_WHHF, K_BIHF, K_BHHF, K_WIHB, K_WHHB, K_BIHB, K_BHHB, K_WOUT, K_BOUT
};

extern "C" void kernel_launch(void* const* d_in, const int* in_sizes, int n_in,
                              void* d_out, int out_size, void* d_ws, size_t ws_size,
                              hipStream_t stream) {
  (void)in_sizes; (void)n_in; (void)out_size; (void)ws_size;

  auto F = [&](int i) { return (const float*)d_in[i]; };

  // workspace carve-up
  char* w = (char*)d_ws;
  size_t off = 0;
  float* f19T = (float*)(w + off);
  off += (size_t)B_ROWS * 19 * sizeof(float);  off = (off + 255) & ~(size_t)255;
  _Float16* h32 = (_Float16*)(w + off);
  off += (size_t)NBR * B_ROWS * 32 * sizeof(_Float16); off = (off + 255) & ~(size_t)255;
  float* partial = (float*)(w + off);

  float* dout = (float*)d_out;

  // ---- 1) one-hot staging (transposed) ----
  f19_kernel<<<(B_ROWS + 255) / 256, 256, 0, stream>>>(F(IN_FEAT), f19T);

  // ---- 2) branch MLPs (WMMA) ----
  BranchMlpParams m7;
  m7.idx = (const int*)d_in[IN_IDX7]; m7.branchBase = 0;
  m7.bn0_g = F(P7_BASE + K_BN0G); m7.bn0_b = F(P7_BASE + K_BN0B);
  m7.bn0_m = F(P7_BASE + K_BN0M); m7.bn0_v = F(P7_BASE + K_BN0V);
  m7.W0 = F(P7_BASE + K_W0); m7.b0 = F(P7_BASE + K_B0);
  m7.W1 = F(P7_BASE + K_W1); m7.b1 = F(P7_BASE + K_B1);
  m7.bn1_g = F(P7_BASE + K_BN1G); m7.bn1_b = F(P7_BASE + K_BN1B);
  m7.bn1_m = F(P7_BASE + K_BN1M); m7.bn1_v = F(P7_BASE + K_BN1V);
  m7.W2 = F(P7_BASE + K_W2); m7.b2 = F(P7_BASE + K_B2);
  m7.bn2_g = F(P7_BASE + K_BN2G); m7.bn2_b = F(P7_BASE + K_BN2B);
  m7.bn2_m = F(P7_BASE + K_BN2M); m7.bn2_v = F(P7_BASE + K_BN2V);

  BranchMlpParams m3 = m7;
  m3.idx = (const int*)d_in[IN_IDX3]; m3.branchBase = N7;
  m3.bn0_g = F(P3_BASE + K_BN0G); m3.bn0_b = F(P3_BASE + K_BN0B);
  m3.bn0_m = F(P3_BASE + K_BN0M); m3.bn0_v = F(P3_BASE + K_BN0V);
  m3.W0 = F(P3_BASE + K_W0); m3.b0 = F(P3_BASE + K_B0);
  m3.W1 = F(P3_BASE + K_W1); m3.b1 = F(P3_BASE + K_B1);
  m3.bn1_g = F(P3_BASE + K_BN1G); m3.bn1_b = F(P3_BASE + K_BN1B);
  m3.bn1_m = F(P3_BASE + K_BN1M); m3.bn1_v = F(P3_BASE + K_BN1V);
  m3.W2 = F(P3_BASE + K_W2); m3.b2 = F(P3_BASE + K_B2);
  m3.bn2_g = F(P3_BASE + K_BN2G); m3.bn2_b = F(P3_BASE + K_BN2B);
  m3.bn2_m = F(P3_BASE + K_BN2M); m3.bn2_v = F(P3_BASE + K_BN2V);

  branch_mlp_kernel<7><<<dim3(MLP_BLOCKS, N7), 32 * MLP_WAVES, 0, stream>>>(m7, f19T, h32);
  branch_mlp_kernel<3><<<dim3(MLP_BLOCKS, N3), 32 * MLP_WAVES, 0, stream>>>(m3, f19T, h32);

  // ---- 3) GRU recurrences ----
  GruParams g7;
  g7.branchBase = 0;
  g7.Wih_f = F(P7_BASE + K_WIHF); g7.Whh_f = F(P7_BASE + K_WHHF);
  g7.bih_f = F(P7_BASE + K_BIHF); g7.bhh_f = F(P7_BASE + K_BHHF);
  g7.Wih_b = F(P7_BASE + K_WIHB); g7.Whh_b = F(P7_BASE + K_WHHB);
  g7.bih_b = F(P7_BASE + K_BIHB); g7.bhh_b = F(P7_BASE + K_BHHB);
  g7.Wout  = F(P7_BASE + K_WOUT);

  GruParams g3;
  g3.branchBase = N7;
  g3.Wih_f = F(P3_BASE + K_WIHF); g3.Whh_f = F(P3_BASE + K_WHHF);
  g3.bih_f = F(P3_BASE + K_BIHF); g3.bhh_f = F(P3_BASE + K_BHHF);
  g3.Wih_b = F(P3_BASE + K_WIHB); g3.Whh_b = F(P3_BASE + K_WHHB);
  g3.bih_b = F(P3_BASE + K_BIHB); g3.bhh_b = F(P3_BASE + K_BHHB);
  g3.Wout  = F(P3_BASE + K_WOUT);

  gru_kernel<<<dim3(2, N7), 32, 0, stream>>>(g7, h32, partial);
  gru_kernel<<<dim3(2, N3), 32, 0, stream>>>(g3, h32, partial);

  // ---- 4) combine to dl_outputs (second output region of d_out) ----
  combine_kernel<<<(B_ROWS * NBR + 255) / 256, 256, 0, stream>>>(
      partial, F(P7_BASE + K_BOUT), F(P3_BASE + K_BOUT), dout);

  // ---- 5) output head -> pred_result (first output region of d_out) ----
  out_kernel<<<(B_ROWS + 255) / 256, 256, 0, stream>>>(
      dout, F(OUT_BASE + 0), F(OUT_BASE + 1), F(OUT_BASE + 2), F(OUT_BASE + 3));
}
